// Decoder_84301618086446
// MI455X (gfx1250) — compile-verified, matrix operand order checked by reading, not run on hardware
//
#include <hip/hip_runtime.h>

#define DEV_INLINE __device__ __forceinline__

typedef __bf16 v16bf __attribute__((ext_vector_type(16)));
typedef __bf16 v8bf  __attribute__((ext_vector_type(8)));
typedef float  v8f   __attribute__((ext_vector_type(8)));
typedef float  v4f   __attribute__((ext_vector_type(4)));
typedef unsigned int u32x4 __attribute__((ext_vector_type(4)));
typedef int    i32x4 __attribute__((ext_vector_type(4)));
typedef int    i32x8 __attribute__((ext_vector_type(8)));

constexpr int S  = 50;
constexpr int B  = 512;
constexpr int E  = 512;
constexpr int H  = 512;
constexpr int FH = 4 * H;
constexpr float C_TANH = 10.0f;
constexpr int NT = 4;  // 16x16 n-tiles per wave (16x64 output per wave)

DEV_INLINE __bf16 f2bf(float f) { return (__bf16)f; }
DEV_INLINE float  sigmoidf_(float x) { return 1.0f / (1.0f + __expf(-x)); }

// CDNA5 16-bit fragment: per lane, vector elements 0..7 hold K = kb+8*half..+7,
// elements 8..15 hold K = kb+16+8*half..+7  -> two contiguous 16B loads.
DEV_INLINE v16bf load_frag_bf(const __bf16* __restrict__ A, int row, int K, int kb, int half) {
    const __bf16* p = A + (size_t)row * K + kb + (half << 3);
    v8bf lo = *(const v8bf*)p;
    v8bf hi = *(const v8bf*)(p + 16);
    v16bf r;
#pragma unroll
    for (int i = 0; i < 8; ++i) { r[i] = lo[i]; r[8 + i] = hi[i]; }
    return r;
}

DEV_INLINE v16bf load_frag_f32(const float* __restrict__ A, int row, int K, int kb, int half) {
    const float* p = A + (size_t)row * K + kb + (half << 3);
    v4f a0 = *(const v4f*)p;
    v4f a1 = *(const v4f*)(p + 4);
    v4f a2 = *(const v4f*)(p + 16);
    v4f a3 = *(const v4f*)(p + 20);
    v16bf r;
#pragma unroll
    for (int i = 0; i < 4; ++i) {
        r[i]      = f2bf(a0[i]);
        r[4 + i]  = f2bf(a1[i]);
        r[8 + i]  = f2bf(a2[i]);
        r[12 + i] = f2bf(a3[i]);
    }
    return r;
}

// C[M,N] = A1[M,K] * W1[N,K]^T (+ A2[M,K] * W2[N,K]^T) (+ bias[N])
// One 16x64 output tile per wave32: A fragment reused across NT=4 WMMAs.
template <bool A1F32, bool A2F32, bool HASA2, bool OUTBF16>
__global__ void gemm_wmma(void* __restrict__ Cout,
                          const void* __restrict__ A1, const __bf16* __restrict__ W1,
                          const void* __restrict__ A2, const __bf16* __restrict__ W2,
                          const float* __restrict__ bias,
                          int M, int N, int K) {
    const int lane    = threadIdx.x & 31;
    const int wave    = threadIdx.x >> 5;
    const int tilesN  = N >> 4;
    const int groupsN = tilesN / NT;
    const int tile    = blockIdx.x * (blockDim.x >> 5) + wave;
    if (tile >= (M >> 4) * groupsN) return;  // wave-uniform exit: EXEC stays all-ones
    const int m0   = (tile / groupsN) << 4;
    const int n0   = (tile % groupsN) * (NT << 4);
    const int half = lane >> 4;
    const int mrow = m0 + (lane & 15);
    const int nlo  = lane & 15;

    v8f acc[NT];
#pragma unroll
    for (int j = 0; j < NT; ++j) acc[j] = (v8f){0.f, 0.f, 0.f, 0.f, 0.f, 0.f, 0.f, 0.f};

    for (int kb = 0; kb < K; kb += 32) {
        __builtin_prefetch((const char*)(W1 + (size_t)(n0 + nlo) * K + kb + 32), 0, 0);
        v16bf a = A1F32 ? load_frag_f32((const float*)A1, mrow, K, kb, half)
                        : load_frag_bf((const __bf16*)A1, mrow, K, kb, half);
#pragma unroll
        for (int j = 0; j < NT; ++j) {
            v16bf b = load_frag_bf(W1, n0 + (j << 4) + nlo, K, kb, half);
            acc[j] = __builtin_amdgcn_wmma_f32_16x16x32_bf16(false, a, false, b,
                                                             (short)0, acc[j], false, false);
        }
    }
    if constexpr (HASA2) {
        for (int kb = 0; kb < K; kb += 32) {
            __builtin_prefetch((const char*)(W2 + (size_t)(n0 + nlo) * K + kb + 32), 0, 0);
            v16bf a = A2F32 ? load_frag_f32((const float*)A2, mrow, K, kb, half)
                            : load_frag_bf((const __bf16*)A2, mrow, K, kb, half);
#pragma unroll
            for (int j = 0; j < NT; ++j) {
                v16bf b = load_frag_bf(W2, n0 + (j << 4) + nlo, K, kb, half);
                acc[j] = __builtin_amdgcn_wmma_f32_16x16x32_bf16(false, a, false, b,
                                                                 (short)0, acc[j], false, false);
            }
        }
    }

#pragma unroll
    for (int j = 0; j < NT; ++j) {
        const int n  = n0 + (j << 4) + nlo;
        const float bv = bias ? bias[n] : 0.0f;
#pragma unroll
        for (int r = 0; r < 8; ++r) {
            // C/D layout: VGPR r -> M = r (lanes 0-15) or r+8 (lanes 16-31), N = lane%16
            int m = m0 + r + (half << 3);
            float val = acc[j][r] + bv;
            if constexpr (OUTBF16)
                ((__bf16*)Cout)[(size_t)m * N + n] = f2bf(val);
            else
                ((float*)Cout)[(size_t)m * N + n] = val;
        }
    }
}

// TDM warm kernel: DMA one 16x512 bf16 weight tile into LDS (Tensor Data Mover),
// wait on TENSORcnt, cluster-barrier (NOP outside a cluster). Output-neutral; warms L2.
__global__ void __launch_bounds__(32) tdm_prefetch_kernel(const __bf16* __restrict__ src,
                                                          int rowsPerTile, int K) {
    __shared__ __align__(128) char lds_buf[16 * 512 * 2];
#if defined(__HIP_DEVICE_COMPILE__)
#if __has_builtin(__builtin_amdgcn_tensor_load_to_lds)
    const unsigned long long tileBytes = (unsigned long long)rowsPerTile * K * 2ull;
    unsigned long long gaddr = (unsigned long long)src + (unsigned long long)blockIdx.x * tileBytes;
#if __clang_major__ >= 23
    // Field semantics of the 6-arg form differ; pass NULL-tensor (count=0) => arch NOP.
    u32x4 g0 = {0u, 0u, 0u, 0u};
    i32x8 g1 = {0, 0, 0, 0, 0, 0, 0, 0};
    i32x4 g2 = {0, 0, 0, 0}, g3 = {0, 0, 0, 0};
    i32x8 g4 = {0, 0, 0, 0, 0, 0, 0, 0};
    __builtin_amdgcn_tensor_load_to_lds(g0, g1, g2, g3, g4, 0);
#else
    // D# group0: count=1 (valid), lds_addr=0, global_addr, type=2 ("image")
    u32x4 g0;
    g0[0] = 1u;
    g0[1] = 0u;
    g0[2] = (unsigned)(gaddr & 0xffffffffu);
    g0[3] = (unsigned)((gaddr >> 32) & 0x1ffffffu) | (2u << 30);
    // D# group1: data_size=1 (2B); tensor = tile = rowsPerTile x K; stride0 = K
    i32x8 g1;
    g1[0] = (1 << 16);                                         // data_size=2B
    g1[1] = (int)(((unsigned)K & 0xffffu) << 16);              // tensor_dim0[15:0]
    g1[2] = (int)((((unsigned)K >> 16) & 0xffffu)              // tensor_dim0[31:16]
                  | (((unsigned)rowsPerTile & 0xffffu) << 16));// tensor_dim1[15:0]
    g1[3] = (int)((((unsigned)rowsPerTile >> 16) & 0xffffu)    // tensor_dim1[31:16]
                  | (((unsigned)K & 0xffffu) << 16));          // tile_dim0
    g1[4] = (int)((unsigned)rowsPerTile & 0xffffu);            // tile_dim1 (tile_dim2=0)
    g1[5] = K;                                                 // tensor_dim0_stride[31:0]
    g1[6] = 0;
    g1[7] = 0;
    i32x4 g2 = {0, 0, 0, 0}, g3 = {0, 0, 0, 0};
    __builtin_amdgcn_tensor_load_to_lds(g0, g1, g2, g3, 0);
#endif
    __builtin_amdgcn_s_wait_tensorcnt(0);
#endif
    __builtin_amdgcn_s_cluster_barrier();
    // keep the LDS buffer alive
    __asm__ volatile("" ::"v"(*(volatile int*)lds_buf));
#endif
}

// elementwise f32 -> bf16
__global__ void f32_to_bf16_kernel(__bf16* __restrict__ dst, const float* __restrict__ src, int n) {
    int i = blockIdx.x * blockDim.x + threadIdx.x;
    if (i < n) dst[i] = f2bf(src[i]);
}

// LSTM pointwise: gates G[B,4H] (i,f,g,o) + biases -> new h,c (in place)
__global__ void lstm_pointwise(const float* __restrict__ G,
                               const float* __restrict__ bih, const float* __restrict__ bhh,
                               float* __restrict__ h, float* __restrict__ c) {
    int i = blockIdx.x * blockDim.x + threadIdx.x;
    if (i >= B * H) return;
    int b = i / H, hh = i % H;
    const float* g = G + (size_t)b * FH;
    float gi = g[hh]         + bih[hh]         + bhh[hh];
    float gf = g[H + hh]     + bih[H + hh]     + bhh[H + hh];
    float gg = g[2 * H + hh] + bih[2 * H + hh] + bhh[2 * H + hh];
    float go = g[3 * H + hh] + bih[3 * H + hh] + bhh[3 * H + hh];
    float c2 = sigmoidf_(gf) * c[i] + sigmoidf_(gi) * tanhf(gg);
    c[i] = c2;
    h[i] = sigmoidf_(go) * tanhf(c2);
}

// out[b,s] = sum_h v[h]*tanh(Q[b,h] + e[s,b,h]); optional C*tanh; mask -> -inf
// one wave32 per (b,s); each lane handles 16 contiguous H elements (b128 loads)
__global__ void attn_logits_kernel(float* __restrict__ out, const float* __restrict__ Q,
                                   const __bf16* __restrict__ e, const float* __restrict__ v,
                                   const unsigned char* __restrict__ mask, float scaleTanh) {
    int wid  = (blockIdx.x * blockDim.x + threadIdx.x) >> 5;
    int lane = threadIdx.x & 31;
    if (wid >= B * S) return;
    int b = wid / S, s = wid % S;
    const float*  qp = Q + (size_t)b * H + lane * 16;
    const float*  vp = v + lane * 16;
    const __bf16* ep = e + ((size_t)s * B + b) * H + lane * 16;
    v8bf e0 = *(const v8bf*)ep;
    v8bf e1 = *(const v8bf*)(ep + 8);
    float sum = 0.0f;
#pragma unroll
    for (int i = 0; i < 8; ++i) sum += vp[i] * tanhf(qp[i] + (float)e0[i]);
#pragma unroll
    for (int i = 0; i < 8; ++i) sum += vp[8 + i] * tanhf(qp[8 + i] + (float)e1[i]);
#pragma unroll
    for (int off = 16; off; off >>= 1) sum += __shfl_xor(sum, off, 32);
    if (lane == 0) {
        float val = (scaleTanh > 0.0f) ? scaleTanh * tanhf(sum) : sum;
        out[wid] = mask[b * S + s] ? -__builtin_inff() : val;
    }
}

// softmax over S per batch row (one block of 64 threads per b)
__global__ void softmax_rows(float* __restrict__ w, const float* __restrict__ logits) {
    int b = blockIdx.x;
    int t = threadIdx.x;
    __shared__ float sm[64];
    float v = (t < S) ? logits[b * S + t] : -__builtin_inff();
    sm[t] = v;
    __syncthreads();
    for (int off = 32; off; off >>= 1) {
        if (t < off) sm[t] = fmaxf(sm[t], sm[t + off]);
        __syncthreads();
    }
    float mx = sm[0];
    __syncthreads();
    float ex = (t < S) ? __expf(v - mx) : 0.0f;
    sm[t] = ex;
    __syncthreads();
    for (int off = 32; off; off >>= 1) {
        if (t < off) sm[t] += sm[t + off];
        __syncthreads();
    }
    float inv = 1.0f / sm[0];
    if (t < S) w[b * S + t] = ex * inv;
}

// g[b,h] = sum_s e[s,b,h] * w[b,s]
__global__ void glimpse_kernel(float* __restrict__ g, const __bf16* __restrict__ e,
                               const float* __restrict__ w) {
    int i = blockIdx.x * blockDim.x + threadIdx.x;
    if (i >= B * H) return;
    int b = i / H, hh = i % H;
    float acc = 0.0f;
    for (int s = 0; s < S; ++s)
        acc += (float)e[((size_t)s * B + b) * H + hh] * w[b * S + s];
    g[i] = acc;
}

// mask0 = mask_modify(V_reach_mask)
__global__ void mask_init(unsigned char* __restrict__ mask, const unsigned char* __restrict__ vmask) {
    int b = blockIdx.x * blockDim.x + threadIdx.x;
    if (b >= B) return;
    bool all = true;
    for (int s = 0; s < S; ++s) {
        unsigned char m = vmask[b * S + s] ? 1 : 0;
        mask[b * S + s] = m;
        if (!m) all = false;
    }
    if (all) mask[b * S + S - 1] = 0;
}

// log_softmax fusion + greedy argmax + mask update + next-input gather (bf16)
__global__ void combine_kernel(float* __restrict__ out_logp, int* __restrict__ out_sel,
                               const float* __restrict__ lp, const float* __restrict__ lpa,
                               unsigned char* __restrict__ mask,
                               const float* __restrict__ emb, const float* __restrict__ enc_aoi,
                               __bf16* __restrict__ x_bf, __bf16* __restrict__ xa_bf, int t) {
    int b  = blockIdx.x;
    int tt = threadIdx.x;  // 128 threads
    __shared__ float slp[64], slpa[64];
    __shared__ int sidx;
    if (tt < S) { slp[tt] = lp[b * S + tt]; slpa[tt] = lpa[b * S + tt]; }
    __syncthreads();
    if (tt == 0) {
        float m1 = -__builtin_inff(), m2 = -__builtin_inff();
        for (int s = 0; s < S; ++s) { m1 = fmaxf(m1, slp[s]); m2 = fmaxf(m2, slpa[s]); }
        float s1 = 0.f, s2 = 0.f;
        for (int s = 0; s < S; ++s) { s1 += __expf(slp[s] - m1); s2 += __expf(slpa[s] - m2); }
        float l1 = m1 + __logf(s1), l2 = m2 + __logf(s2);
        float best = -__builtin_inff(); int bi = 0;
        for (int s = 0; s < S; ++s) {
            float v = (slp[s] - l1) + 0.1f * (slpa[s] - l2);
            slp[s] = v;
            if (v > best) { best = v; bi = s; }
        }
        sidx = bi;
        mask[b * S + bi] = 1;
        bool all = true;
        for (int s = 0; s < S; ++s) if (!mask[b * S + s]) all = false;
        if (all) mask[b * S + S - 1] = 0;
        out_sel[(size_t)b * S + t] = bi;
    }
    __syncthreads();
    if (tt < S) out_logp[(size_t)b * S * S + (size_t)t * S + tt] = slp[tt];
    int bi = sidx;
    for (int ee = tt; ee < E; ee += blockDim.x) {
        x_bf[(size_t)b * E + ee]  = f2bf(emb[((size_t)bi * B + b) * E + ee]);
        xa_bf[(size_t)b * E + ee] = f2bf(enc_aoi[((size_t)bi * B + b) * E + ee]);
    }
}

extern "C" void kernel_launch(void* const* d_in, const int* in_sizes, int n_in,
                              void* d_out, int out_size, void* d_ws, size_t ws_size,
                              hipStream_t stream) {
    (void)in_sizes; (void)n_in; (void)out_size; (void)ws_size;

    const float* decoder_input = (const float*)d_in[0];
    const float* embedded      = (const float*)d_in[1];
    const float* h0   = (const float*)d_in[2];
    const float* c0   = (const float*)d_in[3];
    const float* context    = (const float*)d_in[4];
    const float* enc_h_aoi  = (const float*)d_in[5];
    const float* dec_in_aoi = (const float*)d_in[6];
    const float* h0a  = (const float*)d_in[7];
    const float* c0a  = (const float*)d_in[8];
    const unsigned char* vmask = (const unsigned char*)d_in[9];
    const float* W_ih = (const float*)d_in[10];
    const float* W_hh = (const float*)d_in[11];
    const float* b_ih = (const float*)d_in[12];
    const float* b_hh = (const float*)d_in[13];
    const float* W_ih_a = (const float*)d_in[14];
    const float* W_hh_a = (const float*)d_in[15];
    const float* b_ih_a = (const float*)d_in[16];
    const float* b_hh_a = (const float*)d_in[17];
    const float* Wq_p  = (const float*)d_in[18]; const float* bq_p  = (const float*)d_in[19];
    const float* Wr_p  = (const float*)d_in[20]; const float* br_p  = (const float*)d_in[21];
    const float* v_p   = (const float*)d_in[22];
    const float* Wq_pa = (const float*)d_in[23]; const float* bq_pa = (const float*)d_in[24];
    const float* Wr_pa = (const float*)d_in[25]; const float* br_pa = (const float*)d_in[26];
    const float* v_pa  = (const float*)d_in[27];
    const float* Wq_g  = (const float*)d_in[28]; const float* bq_g  = (const float*)d_in[29];
    const float* Wr_g  = (const float*)d_in[30]; const float* br_g  = (const float*)d_in[31];
    const float* v_g   = (const float*)d_in[32];
    const float* Wq_ga = (const float*)d_in[33]; const float* bq_ga = (const float*)d_in[34];
    const float* Wr_ga = (const float*)d_in[35]; const float* br_ga = (const float*)d_in[36];
    const float* v_ga  = (const float*)d_in[37];

    // ---- workspace bump allocator (resident set ~140MB: fits the 192MB L2) ----
    char* base = (char*)d_ws;
    size_t off = 0;
    auto alloc = [&](size_t bytes) -> void* {
        void* p = base + off;
        off += (bytes + 255) & ~(size_t)255;
        return p;
    };

    __bf16* Wih_bf  = (__bf16*)alloc((size_t)FH * E * 2);
    __bf16* Whh_bf  = (__bf16*)alloc((size_t)FH * H * 2);
    __bf16* Wiha_bf = (__bf16*)alloc((size_t)FH * E * 2);
    __bf16* Whha_bf = (__bf16*)alloc((size_t)FH * H * 2);
    __bf16* Wqp_bf  = (__bf16*)alloc((size_t)H * H * 2);
    __bf16* Wqpa_bf = (__bf16*)alloc((size_t)H * H * 2);
    __bf16* Wqg_bf  = (__bf16*)alloc((size_t)H * H * 2);
    __bf16* Wqga_bf = (__bf16*)alloc((size_t)H * H * 2);
    __bf16* Wrp_bf  = (__bf16*)alloc((size_t)H * H * 2);
    __bf16* Wrpa_bf = (__bf16*)alloc((size_t)H * H * 2);
    __bf16* Wrg_bf  = (__bf16*)alloc((size_t)H * H * 2);
    __bf16* Wrga_bf = (__bf16*)alloc((size_t)H * H * 2);
    __bf16* e_g  = (__bf16*)alloc((size_t)S * B * H * 2);
    __bf16* e_p  = (__bf16*)alloc((size_t)S * B * H * 2);
    __bf16* e_ga = (__bf16*)alloc((size_t)S * B * H * 2);
    __bf16* e_pa = (__bf16*)alloc((size_t)S * B * H * 2);
    float* hbuf  = (float*)alloc((size_t)B * H * 4);
    float* cbuf  = (float*)alloc((size_t)B * H * 4);
    float* habuf = (float*)alloc((size_t)B * H * 4);
    float* cabuf = (float*)alloc((size_t)B * H * 4);
    __bf16* x_bf  = (__bf16*)alloc((size_t)B * E * 2);
    __bf16* xa_bf = (__bf16*)alloc((size_t)B * E * 2);
    float* Gbuf  = (float*)alloc((size_t)B * FH * 4);
    float* Gabuf = (float*)alloc((size_t)B * FH * 4);
    float* Qg  = (float*)alloc((size_t)B * H * 4);
    float* Qga = (float*)alloc((size_t)B * H * 4);
    float* Qp  = (float*)alloc((size_t)B * H * 4);
    float* Qpa = (float*)alloc((size_t)B * H * 4);
    float* gl  = (float*)alloc((size_t)B * H * 4);
    float* gla = (float*)alloc((size_t)B * H * 4);
    float* lg   = (float*)alloc((size_t)B * S * 4);
    float* lga  = (float*)alloc((size_t)B * S * 4);
    float* lpb  = (float*)alloc((size_t)B * S * 4);
    float* lpab = (float*)alloc((size_t)B * S * 4);
    float* wsm  = (float*)alloc((size_t)B * S * 4);
    float* wsma = (float*)alloc((size_t)B * S * 4);
    unsigned char* mask = (unsigned char*)alloc((size_t)B * S);

    float* out_logp = (float*)d_out;
    int*   out_sel  = (int*)((float*)d_out + (size_t)B * S * S);

    // ---- one-time: weight conversion to bf16 ----
    struct CV { const float* src; __bf16* dst; int n; };
    const CV cvs[] = {
        {W_ih, Wih_bf, FH * E}, {W_hh, Whh_bf, FH * H},
        {W_ih_a, Wiha_bf, FH * E}, {W_hh_a, Whha_bf, FH * H},
        {Wq_p, Wqp_bf, H * H}, {Wq_pa, Wqpa_bf, H * H},
        {Wq_g, Wqg_bf, H * H}, {Wq_ga, Wqga_bf, H * H},
        {Wr_p, Wrp_bf, H * H}, {Wr_pa, Wrpa_bf, H * H},
        {Wr_g, Wrg_bf, H * H}, {Wr_ga, Wrga_bf, H * H},
    };
    for (const CV& c : cvs)
        f32_to_bf16_kernel<<<(c.n + 255) / 256, 256, 0, stream>>>(c.dst, c.src, c.n);

    // ---- one-time: TDM warm of the recurrent gate weights (tensor DMA into LDS) ----
    tdm_prefetch_kernel<<<FH / 16, 32, 0, stream>>>(Wih_bf, 16, H);
    tdm_prefetch_kernel<<<FH / 16, 32, 0, stream>>>(Whh_bf, 16, H);
    tdm_prefetch_kernel<<<FH / 16, 32, 0, stream>>>(Wiha_bf, 16, H);
    tdm_prefetch_kernel<<<FH / 16, 32, 0, stream>>>(Whha_bf, 16, H);

    // ---- one-time: hoisted projections e = ctx @ Wr^T + br, stored [S,B,H] bf16 ----
    {
        const int M = S * B, N = H, K = H;
        const int waves = (M / 16) * ((N / 16) / NT);
        dim3 grid(waves / 8), blk(256);
        gemm_wmma<true, false, false, true><<<grid, blk, 0, stream>>>(
            e_g, context, Wrg_bf, nullptr, nullptr, br_g, M, N, K);
        gemm_wmma<true, false, false, true><<<grid, blk, 0, stream>>>(
            e_p, context, Wrp_bf, nullptr, nullptr, br_p, M, N, K);
        gemm_wmma<true, false, false, true><<<grid, blk, 0, stream>>>(
            e_ga, dec_in_aoi, Wrga_bf, nullptr, nullptr, br_ga, M, N, K);
        gemm_wmma<true, false, false, true><<<grid, blk, 0, stream>>>(
            e_pa, dec_in_aoi, Wrpa_bf, nullptr, nullptr, br_pa, M, N, K);
    }

    // ---- one-time: initial state ----
    hipMemcpyAsync(hbuf, h0, (size_t)B * H * 4, hipMemcpyDeviceToDevice, stream);
    hipMemcpyAsync(cbuf, c0, (size_t)B * H * 4, hipMemcpyDeviceToDevice, stream);
    hipMemcpyAsync(habuf, h0a, (size_t)B * H * 4, hipMemcpyDeviceToDevice, stream);
    hipMemcpyAsync(cabuf, c0a, (size_t)B * H * 4, hipMemcpyDeviceToDevice, stream);
    f32_to_bf16_kernel<<<(B * E + 255) / 256, 256, 0, stream>>>(x_bf, decoder_input, B * E);
    f32_to_bf16_kernel<<<(B * E + 255) / 256, 256, 0, stream>>>(xa_bf, decoder_input, B * E);
    mask_init<<<(B + 255) / 256, 256, 0, stream>>>(mask, vmask);

    const dim3 gGates(((B / 16) * ((FH / 16) / NT)) / 8);  // 1024 waves -> 128 blocks
    const dim3 gQ(((B / 16) * ((H / 16) / NT)) / 8);       // 256 waves  -> 32 blocks
    const dim3 gBH((B * H) / 256), gLog((B * S) / 8);

    // ---- sequential decode: 50 steps ----
    for (int t = 0; t < S; ++t) {
        gemm_wmma<false, true, true, false><<<gGates, 256, 0, stream>>>(
            Gbuf, x_bf, Wih_bf, hbuf, Whh_bf, nullptr, B, FH, H);
        gemm_wmma<false, true, true, false><<<gGates, 256, 0, stream>>>(
            Gabuf, xa_bf, Wiha_bf, habuf, Whha_bf, nullptr, B, FH, H);
        lstm_pointwise<<<gBH, 256, 0, stream>>>(Gbuf, b_ih, b_hh, hbuf, cbuf);
        lstm_pointwise<<<gBH, 256, 0, stream>>>(Gabuf, b_ih_a, b_hh_a, habuf, cabuf);

        gemm_wmma<true, false, false, false><<<gQ, 256, 0, stream>>>(
            Qg, hbuf, Wqg_bf, nullptr, nullptr, bq_g, B, H, H);
        gemm_wmma<true, false, false, false><<<gQ, 256, 0, stream>>>(
            Qga, habuf, Wqga_bf, nullptr, nullptr, bq_ga, B, H, H);
        attn_logits_kernel<<<gLog, 256, 0, stream>>>(lg, Qg, e_g, v_g, mask, 0.0f);
        attn_logits_kernel<<<gLog, 256, 0, stream>>>(lga, Qga, e_ga, v_ga, mask, 0.0f);
        softmax_rows<<<B, 64, 0, stream>>>(wsm, lg);
        softmax_rows<<<B, 64, 0, stream>>>(wsma, lga);
        glimpse_kernel<<<gBH, 256, 0, stream>>>(gl, e_g, wsm);
        glimpse_kernel<<<gBH, 256, 0, stream>>>(gla, e_ga, wsma);

        gemm_wmma<true, false, false, false><<<gQ, 256, 0, stream>>>(
            Qp, gl, Wqp_bf, nullptr, nullptr, bq_p, B, H, H);
        gemm_wmma<true, false, false, false><<<gQ, 256, 0, stream>>>(
            Qpa, gla, Wqpa_bf, nullptr, nullptr, bq_pa, B, H, H);
        attn_logits_kernel<<<gLog, 256, 0, stream>>>(lpb, Qp, e_p, v_p, mask, C_TANH);
        attn_logits_kernel<<<gLog, 256, 0, stream>>>(lpab, Qpa, e_pa, v_pa, mask, C_TANH);

        combine_kernel<<<B, 128, 0, stream>>>(out_logp, out_sel, lpb, lpab, mask,
                                              embedded, enc_h_aoi, x_bf, xa_bf, t);
    }
}
